// FlashAttentionMPS_90804198572687
// MI455X (gfx1250) — matrix operand order repaired and checked by reading, compiled, not run on hardware
//
#include <hip/hip_runtime.h>
#include <hip/hip_bf16.h>

// ---------------------------------------------------------------------------
// Block-diagonal causal attention for MI455X (gfx1250), wave32 + WMMA f16.
//   B=4, S=4096, E=1024, H=16, BS=128, DH=64, nb=32
// Pipeline: f32->f16 casts -> QKV GEMM (LDS-tiled wmma, async-to-LDS staging)
//           -> blockwise attention (wmma) -> output GEMM (wmma, f32 + bias)
// ---------------------------------------------------------------------------

typedef _Float16 f16;
typedef __attribute__((ext_vector_type(16))) _Float16 v16h;
typedef __attribute__((ext_vector_type(8)))  _Float16 v8h;
typedef __attribute__((ext_vector_type(8)))  float    v8f;

#define B_  4
#define S_  4096
#define E_  1024
#define H_  16
#define DH_ 64
#define BS_ 128
#define NB_ 32
#define SD_ 3072   // 3*E, qkv row stride in halves
#define M_  16384  // B*S

#define KT_  32    // K-step per LDS stage (one wmma k-depth)
#define LDA_ 40    // padded LDS row stride in halves (80B, 16B-aligned, bank-spread)

// ---------------------------------------------------------------------------
// CDNA5 async copy global -> LDS (ASYNCcnt path, bypasses VGPRs)
// ---------------------------------------------------------------------------
__device__ __forceinline__ void async_b128(const f16* g, f16* l) {
  // low 32 bits of a generic pointer to __shared__ == LDS byte address
  unsigned int lds = (unsigned int)(unsigned long long)l;
  asm volatile("global_load_async_to_lds_b128 %0, %1, off"
               :: "v"(lds), "v"(g) : "memory");
}
__device__ __forceinline__ void wait_async() {
  asm volatile("s_wait_asynccnt 0x0" ::: "memory");
}

// ---------------------------------------------------------------------------
// WMMA helpers (CDNA5 ISA 7.12.2 layouts)
// ---------------------------------------------------------------------------
__device__ __forceinline__ v8f wmma16x16x32(v16h a, v16h b, v8f c) {
  return __builtin_amdgcn_wmma_f32_16x16x32_f16(false, a, false, b, (short)0, c,
                                                false, false);
}

// A fragment: 16x32 f16, row-major source, p0 at (row0, k0), ld in halves.
__device__ __forceinline__ v16h load_a_frag(const f16* p0, int ld, int lane) {
  const int r  = lane & 15;
  const int hi = lane >> 4;
  const f16* p = p0 + r * ld + hi * 8;
  union { v16h v; v8h h[2]; } u;
  u.h[0] = *reinterpret_cast<const v8h*>(p);        // K = hi*8 .. hi*8+7
  u.h[1] = *reinterpret_cast<const v8h*>(p + 16);   // K = 16+hi*8 ..
  return u.v;
}

// B fragment: 32x16 f16; column n of B contiguous over K at p0 + n*ld.
__device__ __forceinline__ v16h load_b_frag(const f16* p0, int ld, int lane) {
  const int n  = lane & 15;
  const int hi = lane >> 4;
  const f16* p = p0 + n * ld + hi * 8;
  union { v16h v; v8h h[2]; } u;
  u.h[0] = *reinterpret_cast<const v8h*>(p);
  u.h[1] = *reinterpret_cast<const v8h*>(p + 16);
  return u.v;
}

// ---------------------------------------------------------------------------
// f32 -> f16 cast
// ---------------------------------------------------------------------------
__global__ void cvt_f32_f16(const float* __restrict__ in, f16* __restrict__ out,
                            size_t n) {
  size_t i = (size_t)blockIdx.x * blockDim.x + threadIdx.x;
  if (i < n) out[i] = (f16)in[i];
}

// ---------------------------------------------------------------------------
// GEMM: C[M,N] = A[M,K] * W[N,K]^T, 128x128 workgroup tile, double-buffered
// async-to-LDS staging. 8 waves in 4x2; each wave: 32x64 tile, 8 wmma/K-step.
// ---------------------------------------------------------------------------
template <bool WRITE_F16>
__global__ __launch_bounds__(256)
void gemm_wmma_lds(const f16* __restrict__ A, const f16* __restrict__ W,
                   const float* __restrict__ bias,
                   f16* __restrict__ out16, float* __restrict__ out32,
                   int M, int N, int K) {
  __shared__ __align__(16) f16 As[2][128 * LDA_];   // 2 x 10 KB
  __shared__ __align__(16) f16 Bs[2][128 * LDA_];   // 2 x 10 KB

  const int tid   = threadIdx.x;
  const int lane  = tid & 31;
  const int wave  = tid >> 5;
  const int tM    = blockIdx.x;          // M / 128
  const int tN    = blockIdx.y;          // N / 128
  const int waveM = wave >> 1;           // 0..3, 32 rows each
  const int waveN = wave & 1;            // 0..1, 64 cols each

  const f16* Abase = A + (size_t)(tM * 128) * K;
  const f16* Bbase = W + (size_t)(tN * 128) * K;

  // each thread stages 2 x 16B chunks per tile: rows r0 and r0+64
  const int r0 = tid >> 2;
  const int c0 = (tid & 3) * 8;

  auto stage = [&](int buf, int k0) {
    async_b128(Abase + (size_t)r0 * K + k0 + c0,        &As[buf][r0 * LDA_ + c0]);
    async_b128(Abase + (size_t)(r0 + 64) * K + k0 + c0, &As[buf][(r0 + 64) * LDA_ + c0]);
    async_b128(Bbase + (size_t)r0 * K + k0 + c0,        &Bs[buf][r0 * LDA_ + c0]);
    async_b128(Bbase + (size_t)(r0 + 64) * K + k0 + c0, &Bs[buf][(r0 + 64) * LDA_ + c0]);
  };

  v8f acc[2][4] = {};

  stage(0, 0);
  wait_async();
  __syncthreads();

  int buf = 0;
  for (int k0 = 0; k0 < K; k0 += KT_) {
    if (k0 + KT_ < K) stage(buf ^ 1, k0 + KT_);   // prefetch next stage

    v16h af0 = load_a_frag(&As[buf][(waveM * 32 + 0)  * LDA_], LDA_, lane);
    v16h af1 = load_a_frag(&As[buf][(waveM * 32 + 16) * LDA_], LDA_, lane);
#pragma unroll
    for (int t = 0; t < 4; ++t) {
      v16h bf = load_b_frag(&Bs[buf][(waveN * 64 + t * 16) * LDA_], LDA_, lane);
      acc[0][t] = wmma16x16x32(af0, bf, acc[0][t]);
      acc[1][t] = wmma16x16x32(af1, bf, acc[1][t]);
    }

    wait_async();        // next stage landed (this wave's copies)
    __syncthreads();     // everyone done reading buf, everyone sees buf^1
    buf ^= 1;
  }

  const int hi = lane >> 4;
  const int nn = lane & 15;
#pragma unroll
  for (int m = 0; m < 2; ++m) {
#pragma unroll
    for (int t = 0; t < 4; ++t) {
#pragma unroll
      for (int j = 0; j < 8; ++j) {
        const int row = tM * 128 + waveM * 32 + m * 16 + j + 8 * hi;
        const int col = tN * 128 + waveN * 64 + t * 16 + nn;
        if (WRITE_F16) {
          out16[(size_t)row * N + col] = (f16)acc[m][t][j];
        } else {
          out32[(size_t)row * N + col] = acc[m][t][j] + bias[col];
        }
      }
    }
  }
}

// ---------------------------------------------------------------------------
// Blockwise causal attention: one workgroup per (b, seq-block, head).
// 8 waves; wave w owns score rows [w*16, w*16+16) of the 128x128 block.
// ---------------------------------------------------------------------------
__global__ __launch_bounds__(256)
void attn_block(const f16* __restrict__ qkv, f16* __restrict__ outh) {
  __shared__ __align__(16) f16 VT[DH_ * BS_];   // V^T: 64 x 128, 16 KB
  __shared__ __align__(16) f16 P[BS_ * BS_];    // probabilities, 32 KB

  const int b    = blockIdx.x >> 9;        // / (NB_*H_)
  const int rem  = blockIdx.x & 511;
  const int nblk = rem >> 4;               // / H_
  const int h    = rem & 15;
  const int s0   = nblk * BS_;

  const f16* base = qkv + ((size_t)b * S_ + s0) * SD_;
  const f16* Qb = base + h * DH_;
  const f16* Kb = base + E_ + h * DH_;
  const f16* Vb = base + 2 * E_ + h * DH_;

  // Stage V transposed into LDS (coalesced global reads along dh).
  for (int idx = threadIdx.x; idx < BS_ * DH_; idx += 256) {
    const int k = idx >> 6;
    const int d = idx & 63;
    VT[d * BS_ + k] = Vb[(size_t)k * SD_ + d];
  }
  __syncthreads();

  const int w    = threadIdx.x >> 5;
  const int lane = threadIdx.x & 31;
  const int hi   = lane >> 4;
  const int nn   = lane & 15;

  // ---- scores: S = Q K^T * 1/sqrt(DH), 16x128 strip per wave ----
  v8f sc[8] = {};
  {
    const f16* qrow = Qb + (size_t)w * 16 * SD_;
    v16h a0 = load_a_frag(qrow + 0,  SD_, lane);
    v16h a1 = load_a_frag(qrow + 32, SD_, lane);
#pragma unroll
    for (int t = 0; t < 8; ++t) {
      const f16* krow = Kb + (size_t)t * 16 * SD_;
      v16h b0 = load_b_frag(krow + 0,  SD_, lane);
      v16h b1 = load_b_frag(krow + 32, SD_, lane);
      sc[t] = wmma16x16x32(a0, b0, sc[t]);
      sc[t] = wmma16x16x32(a1, b1, sc[t]);
    }
  }

  // ---- causal mask + softmax (each row lives in one 16-lane half) ----
  const float scale = 0.125f;              // 1/sqrt(64)
  const float NEG = -3.0e38f;
#pragma unroll
  for (int j = 0; j < 8; ++j) {
    const int r = w * 16 + j + 8 * hi;
    float m = NEG;
#pragma unroll
    for (int t = 0; t < 8; ++t) {
      const int c = t * 16 + nn;
      float v = (c <= r) ? (float)sc[t][j] * scale : NEG;
      sc[t][j] = v;
      m = fmaxf(m, v);
    }
#pragma unroll
    for (int off = 1; off < 16; off <<= 1)
      m = fmaxf(m, __shfl_xor(m, off, 32));
    float s = 0.f;
#pragma unroll
    for (int t = 0; t < 8; ++t) {
      float e = __expf((float)sc[t][j] - m);
      sc[t][j] = e;
      s += e;
    }
#pragma unroll
    for (int off = 1; off < 16; off <<= 1)
      s += __shfl_xor(s, off, 32);
    const float inv = 1.0f / s;
#pragma unroll
    for (int t = 0; t < 8; ++t)
      P[r * BS_ + t * 16 + nn] = (f16)((float)sc[t][j] * inv);
  }
  __syncthreads();

  // ---- O = P V : 16x64 strip per wave, K = 128 ----
  v8f oc[4] = {};
  for (int kk = 0; kk < BS_; kk += 32) {
    v16h af = load_a_frag(&P[w * 16 * BS_ + kk], BS_, lane);
#pragma unroll
    for (int t = 0; t < 4; ++t) {
      v16h bf = load_b_frag(&VT[(t * 16) * BS_ + kk], BS_, lane);
      oc[t] = wmma16x16x32(af, bf, oc[t]);
    }
  }

#pragma unroll
  for (int t = 0; t < 4; ++t) {
#pragma unroll
    for (int j = 0; j < 8; ++j) {
      const int row = s0 + w * 16 + j + 8 * hi;
      outh[((size_t)b * S_ + row) * E_ + h * DH_ + t * 16 + nn] = (f16)oc[t][j];
    }
  }
}

// ---------------------------------------------------------------------------
// Launch
// ---------------------------------------------------------------------------
extern "C" void kernel_launch(void* const* d_in, const int* in_sizes, int n_in,
                              void* d_out, int out_size, void* d_ws, size_t ws_size,
                              hipStream_t stream) {
  const float* x     = (const float*)d_in[0];   // [B,S,E]
  const float* w_qkv = (const float*)d_in[1];   // [3E,E]
  const float* w_out = (const float*)d_in[2];   // [E,E]
  const float* b_out = (const float*)d_in[3];   // [E]
  float* out = (float*)d_out;                   // [B,S,E]

  char* ws = (char*)d_ws;
  f16* x16    = (f16*)(ws);                               //  32 MB
  f16* wqkv16 = (f16*)(ws + (size_t)33554432);            //   6 MB
  f16* wout16 = (f16*)(ws + (size_t)39845888);            //   2 MB
  f16* qkv16  = (f16*)(ws + (size_t)41943040);            //  96 MB
  f16* attn16 = (f16*)(ws + (size_t)142606336);           //  32 MB

  const size_t nx = (size_t)M_ * E_;
  const size_t nq = (size_t)3 * E_ * E_;
  const size_t no = (size_t)E_ * E_;
  cvt_f32_f16<<<(unsigned)((nx + 255) / 256), 256, 0, stream>>>(x, x16, nx);
  cvt_f32_f16<<<(unsigned)((nq + 255) / 256), 256, 0, stream>>>(w_qkv, wqkv16, nq);
  cvt_f32_f16<<<(unsigned)((no + 255) / 256), 256, 0, stream>>>(w_out, wout16, no);

  // QKV projection: [16384,3072] = x16 * wqkv16^T
  gemm_wmma_lds<true><<<dim3(M_ / 128, SD_ / 128), 256, 0, stream>>>(
      x16, wqkv16, nullptr, qkv16, nullptr, M_, SD_, E_);

  // blockwise attention: B*nb*H = 2048 workgroups
  attn_block<<<2048, 256, 0, stream>>>(qkv16, attn16);

  // output projection: [16384,1024] = attn16 * wout16^T + bias, f32 out
  gemm_wmma_lds<false><<<dim3(M_ / 128, E_ / 128), 256, 0, stream>>>(
      attn16, wout16, b_out, nullptr, out, M_, E_, E_);
}
